// ChimeraSurrogateNCA_76965813944636
// MI455X (gfx1250) — matrix operand order repaired and checked by reading, not compiled, not run on hardware
//
#include <hip/hip_runtime.h>

typedef __attribute__((ext_vector_type(16))) _Float16 v16h;
typedef __attribute__((ext_vector_type(8)))  _Float16 v8h;
typedef __attribute__((ext_vector_type(8)))  float    v8f;

#define NB     8
#define NC     256
#define NH     64
#define NW     64
#define NPIX   (NB*NH*NW)      /* 32768 */
#define NTAP   9
#define NCB    8               /* channel blocks of 32 */
#define NCHUNK (NCB*NTAP)      /* 72 K-chunks of 32 */
#define STEPS  5
#define EPSV   1e-6f
#define ZPAD   512             /* zero pad bytes appended to each state buffer */
#define XCL_BYTES ((size_t)NPIX*NC*2)

// ---- pack weight[256][256][3][3] f32 -> A-fragment-layout f16 ----
// Chunk k-order: k = (cb*9 + tap)*32 + ci  (must match B build below).
// A 16x32 f16 fragment per lane: row m = lane&15; lane<16 holds K={0..15},
// lane>=16 holds K={16..31}; element e -> K = 16*(lane>>4) + e.
__global__ void hilbert_pack_w(const float* __restrict__ w, _Float16* __restrict__ apack) {
  const int blk   = blockIdx.x;          // 16 mtiles * 72 chunks
  const int mt    = blk / NCHUNK;
  const int chunk = blk % NCHUNK;
  const int cb    = chunk / NTAP;
  const int tap   = chunk % NTAP;
  const int di    = tap / 3, dj = tap % 3;
  const int lane  = threadIdx.x;
  const int m     = mt*16 + (lane & 15);
  const int kh    = lane >> 4;
  v16h f;
#pragma unroll
  for (int e = 0; e < 16; ++e) {
    int k   = 16*kh + e;
    int cin = cb*32 + k;
    f[e] = (_Float16)w[(((m*NC) + cin)*3 + di)*3 + dj];
  }
  *(v16h*)(apack + ((size_t)(mt*NCHUNK + chunk)*32 + lane)*16) = f;
}

// ---- transpose f32 [B,C,H,W] -> f16 channels-last [B,H,W,C] ----
__global__ void hilbert_to_cl(const float* __restrict__ x, _Float16* __restrict__ xcl) {
  size_t i = (size_t)blockIdx.x * blockDim.x + threadIdx.x;
  int xw = (int)(i & 63);
  size_t t = i >> 6;
  int y = (int)(t & 63); t >>= 6;
  int c = (int)(t & 255);
  int b = (int)(t >> 8);
  xcl[((size_t)((b*NH + y)*NW + xw))*NC + c] = (_Float16)x[i];
}

// ---- re-zero the masked-tap landing pads (must run every call) ----
__global__ void hilbert_zero_pad(_Float16* __restrict__ pA, _Float16* __restrict__ pB) {
  const int t = threadIdx.x;          // 256 threads x 2B = 512B per pad
  pA[t] = (_Float16)0.f;
  pB[t] = (_Float16)0.f;
}

// ---- fused step: masked-conv GEMM (WMMA) + squash + RMS norm ----
// Block = 8 waves = M:256 x N:32 pixels (one half-row). Wave w owns M-tiles {2w,2w+1}.
// Masked/out-of-bounds taps are handled by pointing the B-fragment load at the
// zero pad -> no per-dword selects, pure load+wmma inner loop.
__global__ __launch_bounds__(256) void hilbert_step(
    const _Float16* __restrict__ xcur,
    _Float16* __restrict__ xnext,
    const _Float16* __restrict__ apack,
    const float* __restrict__ mask,
    float* __restrict__ out,
    int last)
{
  const int tid  = threadIdx.x;
  const int wave = tid >> 5;
  const int lane = tid & 31;
  const int half = lane >> 4;
  const int ln16 = lane & 15;

  const int idx0 = blockIdx.x * 32;      // flat (b,y,x) pixel base
  const int xw0  = idx0 & 63;            // 0 or 32
  const int y    = (idx0 >> 6) & 63;
  const int b    = idx0 >> 12;

  // Per-lane B-fragment byte offsets for both n-tiles. Masked taps -> zero pad.
  const unsigned zoff  = (unsigned)XCL_BYTES;    // zero pad offset within buffer
  const unsigned koffB = (unsigned)half * 32u;   // lane-half K base, bytes
  unsigned boff[2][NTAP];
#pragma unroll
  for (int nt = 0; nt < 2; ++nt) {
    const int n  = nt*16 + ln16;
    const int px = xw0 + n;
#pragma unroll
    for (int tap = 0; tap < NTAP; ++tap) {
      const int di = tap/3, dj = tap%3;
      int yy = y + di - 1;
      int xx = px + dj - 1;
      int ok = (yy >= 0) & (yy < NH) & (xx >= 0) & (xx < NW);
      if (ok && mask[tap*4096 + y*64 + px] != 0.0f)
        boff[nt][tap] = (unsigned)(((b*NH + yy)*NW + xx) * NC * 2) + koffB;
      else
        boff[nt][tap] = zoff;
    }
  }

  const char* bp = (const char*)xcur;
  const char* ap = (const char*)apack;
  const unsigned aoff0 = (unsigned)((2*wave  )*NCHUNK*32 + lane) * 32u;
  const unsigned aoff1 = (unsigned)((2*wave+1)*NCHUNK*32 + lane) * 32u;

  v8f acc0 = {}, acc1 = {}, acc2 = {}, acc3 = {};

  for (int cb = 0; cb < NCB; ++cb) {
    const unsigned coff = (unsigned)cb * 64u;   // 32 channels * 2B
#pragma unroll
    for (int tap = 0; tap < NTAP; ++tap) {
      const int chunk = cb*NTAP + tap;
      v16h a0 = *(const v16h*)(ap + aoff0 + (unsigned)chunk*1024u);
      v16h a1 = *(const v16h*)(ap + aoff1 + (unsigned)chunk*1024u);
      v16h b0 = *(const v16h*)(bp + boff[0][tap] + coff);
      v16h b1 = *(const v16h*)(bp + boff[1][tap] + coff);
      acc0 = __builtin_amdgcn_wmma_f32_16x16x32_f16(false, a0, false, b0, (short)0, acc0, false, false);
      acc1 = __builtin_amdgcn_wmma_f32_16x16x32_f16(false, a0, false, b1, (short)0, acc1, false, false);
      acc2 = __builtin_amdgcn_wmma_f32_16x16x32_f16(false, a1, false, b0, (short)0, acc2, false, false);
      acc3 = __builtin_amdgcn_wmma_f32_16x16x32_f16(false, a1, false, b1, (short)0, acc3, false, false);
    }
  }

  // ---- epilogue: x += d/(1+|d|), RMS over 256 channels per pixel ----
  // C/D layout: VGPR v, lane<16 -> M=v, lane>=16 -> M=v+8  => per-lane 8 consecutive channels.
  __shared__ float part[2][8][32];
  __shared__ float invr[32];

  float xn[4][8];
  float ssA = 0.f, ssB = 0.f;
#pragma unroll
  for (int a = 0; a < 4; ++a) {
    const int mt = 2*wave + (a >> 1);
    const int nt = a & 1;
    const int p  = nt*16 + ln16;
    const size_t po = ((size_t)((b*NH + y)*NW + xw0 + p))*NC + mt*16 + 8*half;
    v8h xo = *(const v8h*)(xcur + po);
    const v8f d = (a==0) ? acc0 : ((a==1) ? acc1 : ((a==2) ? acc2 : acc3));
    float ss = 0.f;
#pragma unroll
    for (int v = 0; v < 8; ++v) {
      float dd = d[v];
      float s  = dd / (1.0f + fabsf(dd));
      float xv = (float)xo[v] + s;
      xn[a][v] = xv;
      ss += xv*xv;
    }
    if (nt) ssB += ss; else ssA += ss;
  }
  part[0][wave][lane] = ssA;
  part[1][wave][lane] = ssB;
  __syncthreads();
  if (tid < 32) {                       // deterministic fixed-order reduction
    const int nt = tid >> 4;
    const int p  = tid & 15;
    float s = 0.f;
#pragma unroll
    for (int w2 = 0; w2 < 8; ++w2)
      s += part[nt][w2][p] + part[nt][w2][p+16];
    invr[tid] = 1.0f / sqrtf(s * (1.0f/256.0f) + EPSV);
  }
  __syncthreads();

#pragma unroll
  for (int a = 0; a < 4; ++a) {
    const int mt = 2*wave + (a >> 1);
    const int nt = a & 1;
    const int p  = nt*16 + ln16;
    const float inv = invr[p];
    const int ch = mt*16 + 8*half;
    if (!last) {
      const size_t po = ((size_t)((b*NH + y)*NW + xw0 + p))*NC + ch;
      v8h o;
#pragma unroll
      for (int v = 0; v < 8; ++v) o[v] = (_Float16)(xn[a][v]*inv);
      *(v8h*)(xnext + po) = o;
    } else {
#pragma unroll
      for (int v = 0; v < 8; ++v)
        out[(((size_t)b*NC + ch + v)*NH + y)*NW + (xw0 + p)] = xn[a][v]*inv;
    }
  }
}

extern "C" void kernel_launch(void* const* d_in, const int* in_sizes, int n_in,
                              void* d_out, int out_size, void* d_ws, size_t ws_size,
                              hipStream_t stream) {
  (void)in_sizes; (void)n_in; (void)out_size; (void)ws_size;
  const float* retina = (const float*)d_in[0];
  const float* weight = (const float*)d_in[1];
  const float* mask   = (const float*)d_in[2];
  float* out = (float*)d_out;

  char* ws = (char*)d_ws;
  _Float16* apack = (_Float16*)ws;                                   // 1.18 MB
  const size_t apack_bytes = (size_t)16 * NCHUNK * 32 * 16 * 2;
  _Float16* xclA = (_Float16*)(ws + ((apack_bytes + 255) & ~(size_t)255)); // 16.8 MB + pad
  _Float16* xclB = (_Float16*)d_out;  // overlay scratch in d_out; last step reads xclA, writes f32 out

  hilbert_pack_w<<<16*NCHUNK, 32, 0, stream>>>(weight, apack);
  hilbert_to_cl<<<(NB*NC*NH*NW)/256, 256, 0, stream>>>(retina, xclA);
  // zero the 512B masked-tap landing pads of both buffers (d_out one is
  // clobbered by the previous call's final output, so do this every call)
  hilbert_zero_pad<<<1, 256, 0, stream>>>(xclA + NPIX*NC, xclB + NPIX*NC);

  _Float16* cur = xclA;
  _Float16* nxt = xclB;
  for (int s = 0; s < STEPS; ++s) {
    const int last = (s == STEPS - 1);
    hilbert_step<<<NPIX/32, 256, 0, stream>>>(cur, nxt, apack, mask, out, last);
    _Float16* t = cur; cur = nxt; nxt = t;
  }
}
